// ModelD_5179730559633
// MI455X (gfx1250) — matrix operand order repaired and checked by reading, not compiled, not run on hardware
//
#include <hip/hip_runtime.h>
#include <hip/hip_bf16.h>
#include <math.h>

typedef __bf16 bf16_t;
typedef bf16_t v16bf __attribute__((ext_vector_type(16)));
typedef bf16_t v8bf  __attribute__((ext_vector_type(8)));
typedef float  v8f   __attribute__((ext_vector_type(8)));

// K-order for conv2 GEMM: K = tap*8 + c, tap = ky*5 + kx (taps 0..24, padded to 28).
// sh1p: conv1 output, channel-last, spatially padded 32x32 with zero halo of 2.
//   element offset = ((y+ky)*32 + (x+kx))*8 + c   (y,x = true output coords)
// => for a fixed tap, the 8 channels are 16 contiguous, 16B-aligned bytes.
// sh2: conv2 output, pixel-major [784][16] so each lane's 8-channel C-fragment
//   slice stores as one 16B-aligned ds_store_b128.

struct SMem {
    float sx[784];        // input image (fp32, also y_0 of the iteration)
    float sw1[200];       // conv1 weights
    float sb1[8];
    float sb2[16];
    float siter[25];      // iteration conv weights
    float sfcb[10];
    float slog[10];
    float sprob[10];
    float spool[1296];    // pooled features
    alignas(16) bf16_t sh2[784 * 16];            // conv2 output (bf16, pixel-major)
    union {
        struct {                                 // live: conv1 -> conv2
            alignas(16) bf16_t sh1p[32 * 32 * 8];  // 16384 B, zero halo
            alignas(16) bf16_t sw2[16 * 224];      // 7168 B, K-reordered, zero pad
        } p1;
        struct {                                 // live: fc reduction + iteration
            float red[10 * 256];
            float sy[2][784];
        } p2;
    } u;
};

// compile-time LDS element offset of a tap (clamped: fake taps 25..27 -> 24)
__device__ __forceinline__ constexpr int tap_off(int t) {
    int tc = (t > 24) ? 24 : t;
    return ((tc / 5) * 32 + (tc % 5)) * 8;
}

__global__ __launch_bounds__(256) void fused_cnn_router_kernel(
    const float* __restrict__ x,   const float* __restrict__ w1,
    const float* __restrict__ b1,  const float* __restrict__ w2,
    const float* __restrict__ b2,  const float* __restrict__ fcw,
    const float* __restrict__ fcb, const float* __restrict__ iterw,
    float* __restrict__ out)
{
    __shared__ SMem sm;
    const int b   = blockIdx.x;
    const int tid = threadIdx.x;

    // ---- stage inputs/weights into LDS ----
    const float* xb = x + b * 784;
    for (int i = tid; i < 784; i += 256) sm.sx[i] = xb[i];
    for (int i = tid; i < 200; i += 256) sm.sw1[i] = w1[i];
    if (tid < 8)  sm.sb1[tid]   = b1[tid];
    if (tid < 16) sm.sb2[tid]   = b2[tid];
    if (tid < 25) sm.siter[tid] = iterw[tid];
    if (tid < 10) sm.sfcb[tid]  = fcb[tid];
    // zero-fill padded conv1 buffer (halo must be zero)
    for (int i = tid; i < 32 * 32 * 8; i += 256) sm.u.p1.sh1p[i] = (bf16_t)0.0f;
    // conv2 weights, re-permuted to K = tap*8 + c, zero-padded K in [200,224)
    for (int i = tid; i < 16 * 224; i += 256) {
        const int o = i / 224, k = i % 224;
        bf16_t v = (bf16_t)0.0f;
        if (k < 200) {
            const int t = k / 8, c = k % 8;     // tap, in-channel
            v = (bf16_t)w2[o * 200 + c * 25 + t];
        }
        sm.u.p1.sw2[i] = v;
    }
    __syncthreads();

    // ---- conv1 (1->8, 5x5, pad 2) + ReLU -> sh1p (bf16, channel-last, padded) ----
    for (int p = tid; p < 784; p += 256) {
        const int y = p / 28, xx = p % 28;
        bf16_t vals[8];
        #pragma unroll
        for (int c = 0; c < 8; ++c) {
            float acc = sm.sb1[c];
            #pragma unroll
            for (int ky = 0; ky < 5; ++ky) {
                const int iy = y + ky - 2;
                if ((unsigned)iy < 28u) {
                    #pragma unroll
                    for (int kx = 0; kx < 5; ++kx) {
                        const int ix = xx + kx - 2;
                        if ((unsigned)ix < 28u)
                            acc += sm.sw1[c * 25 + ky * 5 + kx] * sm.sx[iy * 28 + ix];
                    }
                }
            }
            vals[c] = (bf16_t)fmaxf(acc, 0.0f);
        }
        *reinterpret_cast<v8bf*>(&sm.u.p1.sh1p[((y + 2) * 32 + (xx + 2)) * 8]) =
            *reinterpret_cast<const v8bf*>(vals);
    }
    __syncthreads();

    // ---- conv2 (8->16, 5x5, pad 2) as GEMM on v_wmma_f32_16x16x32_bf16 ----
    // M=16 out channels, K=224 (7 chunks of 32), N=16-pixel tiles (49 tiles).
    const int  wave  = tid >> 5;
    const int  lane  = tid & 31;
    const bool hivec = (lane >= 16);
    const int  nl    = lane & 15;
    const int  khalf = hivec ? 8 : 0;          // 16-bit A layout K-half split

    // A fragments depend only on the K-chunk: hoist all 7 out of the tile loop.
    const int abase = nl * 224 + khalf;        // lane's weight row (M = nl)
    v16bf afrag[7];
    #pragma unroll
    for (int kc = 0; kc < 7; ++kc) {
        const v8bf alo = *reinterpret_cast<const v8bf*>(&sm.u.p1.sw2[abase + kc * 32]);
        const v8bf ahi = *reinterpret_cast<const v8bf*>(&sm.u.p1.sw2[abase + kc * 32 + 16]);
        #pragma unroll
        for (int i = 0; i < 8; ++i) { afrag[kc][i] = alo[i]; afrag[kc][8 + i] = ahi[i]; }
    }

    for (int tile = wave; tile < 49; tile += 8) {
        const int p0 = tile * 16 + nl;          // this lane's output pixel (B/C column)
        const int py = p0 / 28, px = p0 % 28;
        const int offp8 = (py * 32 + px) * 8;   // per-tile base into sh1p

        v8f acc = {};
        #pragma unroll
        for (int kc = 0; kc < 7; ++kc) {
            // B fragment: element e<8 -> K = kc*32 + (hi?8:0) + e; e>=8 -> K = kc*32 + 16 + (hi?8:0) + e-8;
            // e in [0,8): tap 4kc+{0|1}, channel e; e in [8,16): tap 4kc+{2|3}, channel e-8.
            const int o0 = hivec ? tap_off(4 * kc + 1) : tap_off(4 * kc + 0);
            const int o1 = hivec ? tap_off(4 * kc + 3) : tap_off(4 * kc + 2);
            const v8bf b0 = *reinterpret_cast<const v8bf*>(&sm.u.p1.sh1p[offp8 + o0]);
            const v8bf b1 = *reinterpret_cast<const v8bf*>(&sm.u.p1.sh1p[offp8 + o1]);
            v16bf bm;
            #pragma unroll
            for (int i = 0; i < 8; ++i) { bm[i] = b0[i]; bm[8 + i] = b1[i]; }

            acc = __builtin_amdgcn_wmma_f32_16x16x32_bf16(
                      false, afrag[kc], false, bm, (short)0, acc, false, false);
        }
        // C/D layout: VGPR r -> M = r + (lane>=16 ? 8 : 0), N = nl.
        // Bias + ReLU; 8 consecutive channels -> one 16B ds_store_b128.
        bf16_t ovals[8];
        #pragma unroll
        for (int r = 0; r < 8; ++r) {
            const int oc = r + (hivec ? 8 : 0);
            ovals[r] = (bf16_t)fmaxf(acc[r] + sm.sb2[oc], 0.0f);
        }
        *reinterpret_cast<v8bf*>(&sm.sh2[p0 * 16 + khalf * 2]) =
            *reinterpret_cast<const v8bf*>(ovals);
    }
    __syncthreads();

    // ---- maxpool 3x3 stride 3 (28 -> 9), flatten order c*81 + py*9 + px ----
    for (int i = tid; i < 1296; i += 256) {
        const int c = i / 81, r = i % 81, py = r / 9, px = r % 9;
        float m = -1e30f;
        #pragma unroll
        for (int dy = 0; dy < 3; ++dy)
            #pragma unroll
            for (int dx = 0; dx < 3; ++dx)
                m = fmaxf(m, (float)sm.sh2[((3 * py + dy) * 28 + (3 * px + dx)) * 16 + c]);
        sm.spool[i] = m;
    }
    __syncthreads();

    // ---- fc (1296 -> 10), deterministic tree reduction, then softmax ----
    float pacc[10];
    #pragma unroll
    for (int o = 0; o < 10; ++o) pacc[o] = 0.0f;
    for (int j = tid; j < 1296; j += 256) {
        const float v = sm.spool[j];
        #pragma unroll
        for (int o = 0; o < 10; ++o) pacc[o] += v * fcw[o * 1296 + j];
    }
    #pragma unroll
    for (int o = 0; o < 10; ++o) sm.u.p2.red[o * 256 + tid] = pacc[o];
    __syncthreads();
    if (tid < 10) {
        float s = sm.sfcb[tid];
        for (int t = 0; t < 256; ++t) s += sm.u.p2.red[tid * 256 + t];
        sm.slog[tid] = s;
    }
    __syncthreads();
    if (tid == 0) {
        float mx = sm.slog[0];
        for (int o = 1; o < 10; ++o) mx = fmaxf(mx, sm.slog[o]);
        float e[10], ssum = 0.0f;
        for (int o = 0; o < 10; ++o) { e[o] = expf(sm.slog[o] - mx); ssum += e[o]; }
        const float inv = 1.0f / ssum;
        for (int o = 0; o < 10; ++o) sm.sprob[o] = e[o] * inv;
    }
    __syncthreads();

    // ---- expected value over iterated conv: out = sum_n probs[n] * f^n(x) ----
    float oacc[4];
    #pragma unroll
    for (int i = 0; i < 4; ++i) {
        const int p = tid + 256 * i;
        oacc[i] = (p < 784) ? sm.sprob[0] * sm.sx[p] : 0.0f;
    }
    const float* cur = sm.sx;
    for (int n = 1; n < 10; ++n) {
        float* dst = sm.u.p2.sy[n & 1];
        #pragma unroll
        for (int i = 0; i < 4; ++i) {
            const int p = tid + 256 * i;
            if (p < 784) {
                const int y = p / 28, xx = p % 28;
                float s = 0.0f;
                #pragma unroll
                for (int ky = 0; ky < 5; ++ky) {
                    const int iy = y + ky - 2;
                    if ((unsigned)iy < 28u) {
                        #pragma unroll
                        for (int kx = 0; kx < 5; ++kx) {
                            const int ix = xx + kx - 2;
                            if ((unsigned)ix < 28u)
                                s += sm.siter[ky * 5 + kx] * cur[iy * 28 + ix];
                        }
                    }
                }
                dst[p] = s;
            }
        }
        __syncthreads();
        const float pn = sm.sprob[n];
        #pragma unroll
        for (int i = 0; i < 4; ++i) {
            const int p = tid + 256 * i;
            if (p < 784) oacc[i] += pn * dst[p];
        }
        cur = dst;
        __syncthreads();
    }

    float* ob = out + b * 784;
    #pragma unroll
    for (int i = 0; i < 4; ++i) {
        const int p = tid + 256 * i;
        if (p < 784) ob[p] = oacc[i];
    }
}

extern "C" void kernel_launch(void* const* d_in, const int* in_sizes, int n_in,
                              void* d_out, int out_size, void* d_ws, size_t ws_size,
                              hipStream_t stream) {
    (void)n_in; (void)d_ws; (void)ws_size; (void)out_size;
    const float* x     = (const float*)d_in[0];
    const float* w1    = (const float*)d_in[1];
    const float* b1    = (const float*)d_in[2];
    const float* w2    = (const float*)d_in[3];
    const float* b2    = (const float*)d_in[4];
    const float* fcw   = (const float*)d_in[5];
    const float* fcb   = (const float*)d_in[6];
    const float* iterw = (const float*)d_in[7];
    const int B = in_sizes[0] / (28 * 28);

    fused_cnn_router_kernel<<<B, 256, 0, stream>>>(
        x, w1, b1, w2, b2, fcw, fcb, iterw, (float*)d_out);
}
